// _Attn_3616362463484
// MI455X (gfx1250) — compile-verified, matrix operand order checked by reading, not run on hardware
//
#include <hip/hip_runtime.h>
#include <hip/hip_bf16.h>

typedef __attribute__((ext_vector_type(16))) __bf16 v16bf;
typedef __attribute__((ext_vector_type(8)))  __bf16 v8bf;
typedef __attribute__((ext_vector_type(4)))  __bf16 v4bf;
typedef __attribute__((ext_vector_type(8)))  float  v8f;
typedef __attribute__((ext_vector_type(4)))  unsigned int u32x4;
typedef __attribute__((ext_vector_type(8)))  int i32x8;
typedef __attribute__((ext_vector_type(4)))  int i32x4;

#define EMBED  1024
#define HEADS  16
#define HEAD_D 64
#define BATCH  2
#define SEQ    2048
#define MROWS  (BATCH*SEQ)   // 4096

// ---------------- helpers ----------------

__device__ __forceinline__ v16bf cat8(v8bf lo, v8bf hi) {
  return __builtin_shufflevector(lo, hi, 0,1,2,3,4,5,6,7,8,9,10,11,12,13,14,15);
}

__device__ __forceinline__ v8f wmma_bf16(v16bf a, v16bf b, v8f c) {
  // D = A(16x32 bf16) * B(32x16 bf16) + C(16x16 f32)
  return __builtin_amdgcn_wmma_f32_16x16x32_bf16(
      /*neg_a=*/false, a, /*neg_b=*/false, b,
      /*c_mod=*/(short)0, c, /*reuse_a=*/false, /*reuse_b=*/false);
}

// wave32 16-lane row reductions via ds_swizzle (xor masks stay inside each
// 16-lane half == the WMMA C-layout row group).
#define SWZ(x, m) __int_as_float(__builtin_amdgcn_ds_swizzle(__float_as_int(x), (0x1f | ((m) << 10))))

__device__ __forceinline__ float rowMax16(float v) {
  v = fmaxf(v, SWZ(v, 1)); v = fmaxf(v, SWZ(v, 2));
  v = fmaxf(v, SWZ(v, 4)); v = fmaxf(v, SWZ(v, 8));
  return v;
}
__device__ __forceinline__ float rowSum16(float v) {
  v += SWZ(v, 1); v += SWZ(v, 2); v += SWZ(v, 4); v += SWZ(v, 8);
  return v;
}

// ---------------- Tensor Data Mover: 2-D tile -> LDS ----------------
// D# per CDNA5 ISA. data_size=1 (2-byte bf16).
// tensor dims are set equal to the tile dims (tile is always fully in-bounds
// here) with the real row stride, so no OOB zeroing triggers.
// pad_interval: DWORDs before pad (0=2,1=4,2=8,3=16,4=32,...)
// pad_amount:   pad DWORDs minus 1.
__device__ __forceinline__ void tdm_load_2d_bf16(
    unsigned lds_byte_addr, const void* gaddr,
    unsigned tile_d0, unsigned tile_d1, unsigned row_stride_elems,
    unsigned pad_interval, unsigned pad_amount)
{
  unsigned long long ga = (unsigned long long)gaddr;
  u32x4 g0;
  g0[0] = 1u;                                             // count=1, user descriptor
  g0[1] = lds_byte_addr;                                  // lds_addr
  g0[2] = (unsigned)(ga & 0xffffffffu);                   // global_addr[31:0]
  g0[3] = (unsigned)((ga >> 32) & 0x1ffffffu)             // global_addr[56:32]
        | (2u << 30);                                     // type=2 (image)
  i32x8 g1;
  g1[0] = (int)((1u << 16)                                // data_size=1 -> 2B
              | (1u << 20)                                // pad_enable
              | (pad_interval << 22)
              | (pad_amount << 25));
  g1[1] = (int)((tile_d0 & 0xffffu) << 16);               // tensor_dim0[15:0]
  g1[2] = (int)(((tile_d0 >> 16) & 0xffffu)               // tensor_dim0[31:16]
              | ((tile_d1 & 0xffffu) << 16));             // tensor_dim1[15:0]
  g1[3] = (int)(((tile_d1 >> 16) & 0xffffu)
              | ((tile_d0 & 0xffffu) << 16));             // tile_dim0
  g1[4] = (int)(tile_d1 & 0xffffu);                       // tile_dim1 (tile_dim2=0)
  g1[5] = (int)row_stride_elems;                          // tensor_dim0_stride[31:0]
  g1[6] = 0;
  g1[7] = 0;
  i32x4 z4 = {0, 0, 0, 0};
  i32x8 z8 = {0, 0, 0, 0, 0, 0, 0, 0};
  __builtin_amdgcn_tensor_load_to_lds(g0, g1, z4, z4, z8, 0);
}

__device__ __forceinline__ unsigned lds_off(const void* p) {
  return (unsigned)(unsigned long long)p;   // flat LDS addr: low 32 bits = offset
}

// ---------------- fp32 -> bf16 convert ----------------

__global__ __launch_bounds__(256) void k_f32_to_bf16(const float* __restrict__ src,
                                                     __bf16* __restrict__ dst, int n) {
  int i = (blockIdx.x * 256 + threadIdx.x) * 4;
  if (i < n) {
    float4 f = *(const float4*)(src + i);
    v4bf o;
    o[0] = (__bf16)f.x; o[1] = (__bf16)f.y; o[2] = (__bf16)f.z; o[3] = (__bf16)f.w;
    *(v4bf*)(dst + i) = o;
  }
}

// ---------------- WMMA GEMM: out[M,N] = A[M,K] @ W[N,K]^T + bias ----------------
// Block tile 64(M) x 128(N), 8 waves -> each wave owns a 32x32 tile.
// A/B tiles staged into LDS by the Tensor Data Mover (wave 0), with the DMA
// engine applying the 8-element row pad (32 -> 40 elems = 16+4 DWORDs).

template<int N_DIM, bool OUT_BF16>
__global__ __launch_bounds__(256) void k_gemm_bf16(
    const __bf16* __restrict__ A,    // [M, K] row-major
    const __bf16* __restrict__ W,    // [N, K] row-major (torch Linear weight)
    const float*  __restrict__ bias, // [N]
    void* __restrict__ out,          // [M, N] bf16 or f32
    int M, int K)
{
  __shared__ __attribute__((aligned(16))) __bf16 As[64 * 40];
  __shared__ __attribute__((aligned(16))) __bf16 Bs[128 * 40];
  (void)M;

  const int tid   = threadIdx.x;
  const int lane  = tid & 31;
  const int wid   = tid >> 5;
  const int waveM = wid & 1;
  const int waveN = wid >> 1;
  const int m0 = blockIdx.y * 64;
  const int n0 = blockIdx.x * 128;

  const int half = lane >> 4;
  const int l16  = lane & 15;
  const int kb   = half * 8;

  v8f c[2][2] = {};

  for (int k0 = 0; k0 < K; k0 += 32) {
    if (tid < 32) {   // wave 0 programs the TDM for both tiles
      tdm_load_2d_bf16(lds_off(As), A + (size_t)m0 * K + k0,
                       /*tile_d0=*/32, /*tile_d1=*/64, /*stride=*/K,
                       /*pad_interval=*/3, /*pad_amount=*/3);
      tdm_load_2d_bf16(lds_off(Bs), W + (size_t)n0 * K + k0,
                       32, 128, K, 3, 3);
      __builtin_amdgcn_s_wait_tensorcnt(0);
    } else if (k0 + 32 < K) {   // other waves prefetch next k-tile into GL2
      int r = tid >> 2, ch = tid & 3;
      __builtin_prefetch(A + (size_t)(m0 + (r & 63)) * K + k0 + 32 + ch * 8, 0, 1);
      __builtin_prefetch(W + (size_t)(n0 + (tid & 127)) * K + k0 + 32, 0, 1);
    }
    __syncthreads();

    v16bf af[2], bfrag[2];
#pragma unroll
    for (int i = 0; i < 2; ++i) {
      const __bf16* p = As + (waveM * 32 + i * 16 + l16) * 40 + kb;
      af[i] = cat8(*(const v8bf*)p, *(const v8bf*)(p + 16));
    }
#pragma unroll
    for (int j = 0; j < 2; ++j) {
      const __bf16* p = Bs + (waveN * 32 + j * 16 + l16) * 40 + kb;
      bfrag[j] = cat8(*(const v8bf*)p, *(const v8bf*)(p + 16));
    }
#pragma unroll
    for (int i = 0; i < 2; ++i)
#pragma unroll
      for (int j = 0; j < 2; ++j)
        c[i][j] = wmma_bf16(af[i], bfrag[j], c[i][j]);
    __syncthreads();
  }

  // Epilogue: C layout -> m = r + 8*half (within a 16-tile), n = l16
#pragma unroll
  for (int i = 0; i < 2; ++i) {
    int mbase = m0 + waveM * 32 + i * 16 + half * 8;
#pragma unroll
    for (int j = 0; j < 2; ++j) {
      int n = n0 + waveN * 32 + j * 16 + l16;
      float bv = bias[n];
#pragma unroll
      for (int r = 0; r < 8; ++r) {
        float val = c[i][j][r] + bv;
        size_t idx = (size_t)(mbase + r) * N_DIM + n;
        if (OUT_BF16) ((__bf16*)out)[idx] = (__bf16)val;
        else          ((float*)out)[idx]  = val;
      }
    }
  }
}

// ---------------- Flash attention (causal) ----------------
// qkv: [B*T, 3*EMBED] bf16, row layout comp*EMBED + h*64 + d.
// Block = (b,h) x 128 query rows; wave w owns 16 query rows. Key tiles of 32.
// K tile staged by TDM (row 64 elems = 32 DWORDs, pad 4 DWORDs -> stride 72);
// V tile transposed through VGPRs (TDM cannot transpose 16-bit data).

__global__ __launch_bounds__(256) void k_attn(const __bf16* __restrict__ qkv,
                                              __bf16* __restrict__ o) {
  __shared__ __attribute__((aligned(16))) __bf16 Ks[32 * 72];      // [key][d]
  __shared__ __attribute__((aligned(16))) __bf16 Vt[64 * 40];      // [d][key]
  __shared__ __attribute__((aligned(16))) __bf16 Ps[8 * 16 * 40];  // per-wave P

  const int tid  = threadIdx.x;
  const int lane = tid & 31;
  const int wid  = tid >> 5;
  const int half = lane >> 4, l16 = lane & 15, kb = half * 8;

  const int bh = blockIdx.y;
  const int b  = bh / HEADS, h = bh % HEADS;
  const int qBase = blockIdx.x * 128 + wid * 16;

  // Q fragments (A layout): lane row = l16, chunks over d
  v16bf qf[2];
  {
    const __bf16* qrow = qkv + (size_t)(b * SEQ + qBase + l16) * (3 * EMBED) + h * HEAD_D;
#pragma unroll
    for (int t2 = 0; t2 < 2; ++t2) {
      const __bf16* p = qrow + t2 * 32 + kb;
      qf[t2] = cat8(*(const v8bf*)p, *(const v8bf*)(p + 16));
    }
  }

  v8f oacc[4] = {};
  float m_i[8], l_i[8];
#pragma unroll
  for (int r = 0; r < 8; ++r) { m_i[r] = -3.0e38f; l_i[r] = 0.0f; }

  const int nTiles = (blockIdx.x + 1) * 4;   // causal: keys up to end of q block
  __bf16* Pw = Ps + wid * 16 * 40;

  for (int ti = 0; ti < nTiles; ++ti) {
    const int kt0 = ti * 32;
    const int key = tid >> 3, ch = tid & 7;
    { // V tile: per-thread transpose copy (32 keys x 8 chunks of 8)
      size_t row = (size_t)(b * SEQ + kt0 + key) * (3 * EMBED) + h * HEAD_D + ch * 8;
      v8bf vv = *(const v8bf*)(qkv + row + 2 * EMBED);
#pragma unroll
      for (int e = 0; e < 8; ++e) Vt[(ch * 8 + e) * 40 + key] = vv[e];
    }
    if (tid < 32) {   // wave 0: K tile via Tensor Data Mover
      tdm_load_2d_bf16(lds_off(Ks),
                       qkv + (size_t)(b * SEQ + kt0) * (3 * EMBED) + EMBED + h * HEAD_D,
                       /*tile_d0=*/64, /*tile_d1=*/32, /*stride=*/3 * EMBED,
                       /*pad_interval=*/4, /*pad_amount=*/3);
      __builtin_amdgcn_s_wait_tensorcnt(0);
    } else if (ti + 1 < nTiles) {   // prefetch next K/V tile rows
      const __bf16* nf = qkv + (size_t)(b * SEQ + kt0 + 32 + key) * (3 * EMBED)
                       + h * HEAD_D + ch * 8;
      __builtin_prefetch(nf + EMBED, 0, 1);
      __builtin_prefetch(nf + 2 * EMBED, 0, 1);
    }
    __syncthreads();

    if (kt0 <= qBase + 15) {
      // S = Q K^T (two 16x16 tiles over 32 keys)
      v8f s[2];
#pragma unroll
      for (int sn = 0; sn < 2; ++sn) {
        v8f acc = {};
#pragma unroll
        for (int t2 = 0; t2 < 2; ++t2) {
          const __bf16* p = Ks + (sn * 16 + l16) * 72 + t2 * 32 + kb;
          v16bf kf = cat8(*(const v8bf*)p, *(const v8bf*)(p + 16));
          acc = wmma_bf16(qf[t2], kf, acc);
        }
        s[sn] = acc;
      }
      // causal mask + online softmax (row lives in one 16-lane half)
#pragma unroll
      for (int r = 0; r < 8; ++r) {
        int q = qBase + r + half * 8;
        float v0 = s[0][r] * 0.125f;          // 1/sqrt(64)
        float v1 = s[1][r] * 0.125f;
        if (kt0 + l16      > q) v0 = -3.0e38f;
        if (kt0 + 16 + l16 > q) v1 = -3.0e38f;
        float mx   = rowMax16(fmaxf(v0, v1));
        float mnew = fmaxf(m_i[r], mx);
        float p0 = __expf(v0 - mnew);
        float p1 = __expf(v1 - mnew);
        float rs   = rowSum16(p0 + p1);
        float corr = __expf(m_i[r] - mnew);
        l_i[r] = l_i[r] * corr + rs;
        m_i[r] = mnew;
#pragma unroll
        for (int dt = 0; dt < 4; ++dt) oacc[dt][r] *= corr;
        Pw[(r + half * 8) * 40 + l16]      = (__bf16)p0;  // C-layout -> LDS
        Pw[(r + half * 8) * 40 + 16 + l16] = (__bf16)p1;
      }
      // intra-wave LDS RAW: stores above feed A-fragment loads below
      asm volatile("s_wait_dscnt 0" ::: "memory");

      const __bf16* pp = Pw + l16 * 40 + kb;
      v16bf pf = cat8(*(const v8bf*)pp, *(const v8bf*)(pp + 16));
#pragma unroll
      for (int dt = 0; dt < 4; ++dt) {
        const __bf16* vp = Vt + (dt * 16 + l16) * 40 + kb;
        v16bf vf = cat8(*(const v8bf*)vp, *(const v8bf*)(vp + 16));
        oacc[dt] = wmma_bf16(pf, vf, oacc[dt]);
      }
    }
    __syncthreads();
  }

  // normalize + store [B*T, EMBED] bf16
#pragma unroll
  for (int dt = 0; dt < 4; ++dt) {
    int d = dt * 16 + l16;
#pragma unroll
    for (int r = 0; r < 8; ++r) {
      float val = oacc[dt][r] / l_i[r];
      int t = qBase + r + half * 8;
      o[(size_t)(b * SEQ + t) * EMBED + h * HEAD_D + d] = (__bf16)val;
    }
  }
}

// ---------------- launch ----------------

extern "C" void kernel_launch(void* const* d_in, const int* in_sizes, int n_in,
                              void* d_out, int out_size, void* d_ws, size_t ws_size,
                              hipStream_t stream) {
  const float* x     = (const float*)d_in[0];
  const float* w_qkv = (const float*)d_in[1];
  const float* b_qkv = (const float*)d_in[2];
  const float* w_out = (const float*)d_in[3];
  const float* b_out = (const float*)d_in[4];
  float* out = (float*)d_out;

  char* ws = (char*)d_ws;
  __bf16* xh    = (__bf16*)(ws);                 //  8.0 MB  [4096,1024]
  __bf16* wqkvh = (__bf16*)(ws + 8388608);       //  6.0 MB  [3072,1024]
  __bf16* wouth = (__bf16*)(ws + 14680064);      //  2.0 MB  [1024,1024]
  __bf16* qkvh  = (__bf16*)(ws + 16777216);      // 24.0 MB  [4096,3072]
  __bf16* oh    = (__bf16*)(ws + 41943040);      //  8.0 MB  [4096,1024]

  const int nx  = MROWS * EMBED;       // 4,194,304
  const int nwq = 3 * EMBED * EMBED;   // 3,145,728
  const int nwo = EMBED * EMBED;       // 1,048,576

  k_f32_to_bf16<<<nx  / 1024, 256, 0, stream>>>(x,     xh,    nx);
  k_f32_to_bf16<<<nwq / 1024, 256, 0, stream>>>(w_qkv, wqkvh, nwq);
  k_f32_to_bf16<<<nwo / 1024, 256, 0, stream>>>(w_out, wouth, nwo);

  // qkv = x @ w_qkv^T + b_qkv  -> bf16 [4096, 3072]
  k_gemm_bf16<3 * EMBED, true>
      <<<dim3(3 * EMBED / 128, MROWS / 64), 256, 0, stream>>>(
          xh, wqkvh, b_qkv, qkvh, MROWS, EMBED);

  // causal flash attention -> bf16 [4096, 1024]
  k_attn<<<dim3(SEQ / 128, BATCH * HEADS), 256, 0, stream>>>(qkvh, oh);

  // out = o @ w_out^T + b_out -> f32
  k_gemm_bf16<EMBED, false>
      <<<dim3(EMBED / 128, MROWS / 64), 256, 0, stream>>>(
          oh, wouth, b_out, out, MROWS, EMBED);
}
